// GNNLayer_72232759984222
// MI455X (gfx1250) — matrix operand order repaired and checked by reading, not compiled, not run on hardware
//
#include <hip/hip_runtime.h>

#define N_NODES 100000
#define N_EDGES 3200000
#define DFEAT   128
#define EPW     128            // edges per wave in SpMM
#define LDS_H   136            // halfs per LDS row: 128 + 8 pad (bank-conflict free)

typedef __attribute__((ext_vector_type(16))) _Float16 v16h;
typedef __attribute__((ext_vector_type(8)))  _Float16 v8h;
typedef __attribute__((ext_vector_type(8)))  float    v8f;

// ---------------------------------------------------------------- zero ws ---
__global__ __launch_bounds__(256)
void zero4_kernel(float4* __restrict__ p, int n4) {
    int i = blockIdx.x * blockDim.x + threadIdx.x;
    if (i < n4) p[i] = make_float4(0.f, 0.f, 0.f, 0.f);
}

// ----------------------------------------------- W1/W2 -> f16 (once/call) ---
__global__ __launch_bounds__(256)
void cvt_w_kernel(const float* __restrict__ w1, const float* __restrict__ w2,
                  _Float16* __restrict__ w1h, _Float16* __restrict__ w2h, int n) {
    int i = blockIdx.x * blockDim.x + threadIdx.x;
    if (i < n) {
        w1h[i] = (_Float16)w1[i];
        w2h[i] = (_Float16)w2[i];
    }
}

// ------------------------------------------------------------------ SpMM ----
// y[i,:] += val[e] * x[col[e],:] for row[e]==i.  Rows sorted: each wave
// accumulates runs in registers, flushes with f32 atomics at row changes.
// Lane l owns features [4l, 4l+4).  Gathers hit L2 (features fit in 192MB).
__global__ __launch_bounds__(256)
void spmm_kernel(const int*   __restrict__ row,
                 const int*   __restrict__ col,
                 const float* __restrict__ val,
                 const float* __restrict__ x,
                 float*       __restrict__ y) {
    const int lane = threadIdx.x & 31;
    const int wid  = blockIdx.x * (blockDim.x >> 5) + (threadIdx.x >> 5);
    const int e0   = wid * EPW;
    if (e0 >= N_EDGES) return;
    const int e1 = (e0 + EPW < N_EDGES) ? (e0 + EPW) : N_EDGES;

    float4 acc = make_float4(0.f, 0.f, 0.f, 0.f);
    int cur = row[e0];

    for (int e = e0; e < e1; ++e) {
        const int r = row[e];                    // uniform across the wave
        if (r != cur) {                          // uniform branch
            float* yp = y + (size_t)cur * DFEAT + lane * 4;
            atomicAdd(yp + 0, acc.x);
            atomicAdd(yp + 1, acc.y);
            atomicAdd(yp + 2, acc.z);
            atomicAdd(yp + 3, acc.w);
            acc = make_float4(0.f, 0.f, 0.f, 0.f);
            cur = r;
        }
        const int   c = col[e];
        const float v = val[e];
        const float4 xv = *(reinterpret_cast<const float4*>(x + (size_t)c * DFEAT) + lane);
        acc.x += v * xv.x;  acc.y += v * xv.y;
        acc.z += v * xv.z;  acc.w += v * xv.w;
    }
    float* yp = y + (size_t)cur * DFEAT + lane * 4;
    atomicAdd(yp + 0, acc.x);
    atomicAdd(yp + 1, acc.y);
    atomicAdd(yp + 2, acc.z);
    atomicAdd(yp + 3, acc.w);
}

// ----------------------------------------------------------- inter = Lx*f ---
__global__ __launch_bounds__(256)
void inter_kernel(const float* __restrict__ Lx, const float* __restrict__ feat,
                  float* __restrict__ inter, int n4) {
    int i = blockIdx.x * blockDim.x + threadIdx.x;
    if (i < n4) {
        float4 a = ((const float4*)Lx)[i];
        float4 b = ((const float4*)feat)[i];
        ((float4*)inter)[i] = make_float4(a.x * b.x, a.y * b.y, a.z * b.z, a.w * b.w);
    }
}

// ------------------------------------------------- fused dual-GEMM + bias ---
// out = (Lx + diag.*feat) @ W1^T + b1 + Lxi @ W2^T + b2
// Block = 256 threads = 8 waves; block covers 16 rows x 128 cols; each wave
// computes one 16x16 tile with v_wmma_f32_16x16x32_f16 (K=128 in 4 steps).
// A tiles staged into LDS as f16 (converted once per tile); weights read as
// pre-converted f16, one contiguous 32B v16h fragment per lane.
__global__ __launch_bounds__(256)
void gemm_out_kernel(const float*    __restrict__ Lx,   const float* __restrict__ Lxi,
                     const float*    __restrict__ feat, const float* __restrict__ diag,
                     const _Float16* __restrict__ w1h,  const float* __restrict__ b1,
                     const _Float16* __restrict__ w2h,  const float* __restrict__ b2,
                     float* __restrict__ out) {
    __shared__ __attribute__((aligned(16))) _Float16 sA1[16 * LDS_H]; // L1x tile (f16)
    __shared__ __attribute__((aligned(16))) _Float16 sA2[16 * LDS_H]; // Lxi tile (f16)

    const int tid  = threadIdx.x;
    const int row0 = blockIdx.x * 16;

    // Stage: 256 threads x one 8-half chunk = 16 rows x 128 cols, both tiles.
    {
        const int r  = tid >> 4;            // row in tile
        const int kc = (tid & 15) * 8;      // first feature of this chunk
        const size_t gi = (size_t)(row0 + r) * DFEAT + kc;
        const float4 lx0 = *(const float4*)(Lx  + gi);
        const float4 lx1 = *(const float4*)(Lx  + gi + 4);
        const float4 ft0 = *(const float4*)(feat + gi);
        const float4 ft1 = *(const float4*)(feat + gi + 4);
        const float4 xi0 = *(const float4*)(Lxi + gi);
        const float4 xi1 = *(const float4*)(Lxi + gi + 4);
        const float  dg  = diag[row0 + r];
        v8h h1, h2;
        h1[0] = (_Float16)(lx0.x + dg * ft0.x);
        h1[1] = (_Float16)(lx0.y + dg * ft0.y);
        h1[2] = (_Float16)(lx0.z + dg * ft0.z);
        h1[3] = (_Float16)(lx0.w + dg * ft0.w);
        h1[4] = (_Float16)(lx1.x + dg * ft1.x);
        h1[5] = (_Float16)(lx1.y + dg * ft1.y);
        h1[6] = (_Float16)(lx1.z + dg * ft1.z);
        h1[7] = (_Float16)(lx1.w + dg * ft1.w);
        h2[0] = (_Float16)xi0.x;  h2[1] = (_Float16)xi0.y;
        h2[2] = (_Float16)xi0.z;  h2[3] = (_Float16)xi0.w;
        h2[4] = (_Float16)xi1.x;  h2[5] = (_Float16)xi1.y;
        h2[6] = (_Float16)xi1.z;  h2[7] = (_Float16)xi1.w;
        *(v8h*)(&sA1[r * LDS_H + kc]) = h1;
        *(v8h*)(&sA2[r * LDS_H + kc]) = h2;
    }
    __syncthreads();

    const int lane = tid & 31;
    const int wave = tid >> 5;          // output column tile: cols [16w, 16w+16)
    const int m    = lane & 15;         // A row / C column within tile
    const int hi   = lane >> 4;         // lane half selector
    const int colj = wave * 16 + m;     // global output column (= B column)

    const _Float16* w1p = w1h + (size_t)colj * DFEAT;   // W[j,:] -> B[:,j] of W^T
    const _Float16* w2p = w2h + (size_t)colj * DFEAT;

    v8f acc1 = {}; v8f acc2 = {};
#pragma unroll
    for (int kk = 0; kk < DFEAT; kk += 32) {
        // A 16x32 f16 layout: lane half 'hi' holds K = kk+16*hi+e (elems 0..15),
        // row M = lane&15 (matches B's k order).
        const _Float16* a1p = sA1 + m * LDS_H + kk + hi * 16;
        const _Float16* a2p = sA2 + m * LDS_H + kk + hi * 16;
        const v8h a1lo = *(const v8h*)(a1p);
        const v8h a1hi = *(const v8h*)(a1p + 8);
        const v8h a2lo = *(const v8h*)(a2p);
        const v8h a2hi = *(const v8h*)(a2p + 8);
        const v16h a1 = __builtin_shufflevector(a1lo, a1hi,
                            0,1,2,3,4,5,6,7,8,9,10,11,12,13,14,15);
        const v16h a2 = __builtin_shufflevector(a2lo, a2hi,
                            0,1,2,3,4,5,6,7,8,9,10,11,12,13,14,15);
        // B 32x16 f16 layout: lane holds column N = lane&15,
        // K = kk + 16*hi + e for elems 0..15 -> one contiguous 32B load.
        const v16h bw1 = *(const v16h*)(w1p + kk + hi * 16);
        const v16h bw2 = *(const v16h*)(w2p + kk + hi * 16);

        acc1 = __builtin_amdgcn_wmma_f32_16x16x32_f16(false, a1, false, bw1,
                                                      (short)0, acc1, false, false);
        acc2 = __builtin_amdgcn_wmma_f32_16x16x32_f16(false, a2, false, bw2,
                                                      (short)0, acc2, false, false);
    }

    const float bias = b1[colj] + b2[colj];
    // C/D layout: VGPR r -> M = r + 8*hi, N = lane&15.
#pragma unroll
    for (int r = 0; r < 8; ++r) {
        const int M = r + hi * 8;
        out[(size_t)(row0 + M) * DFEAT + wave * 16 + m] = acc1[r] + acc2[r] + bias;
    }
}

// -------------------------------------------------------------- launcher ----
extern "C" void kernel_launch(void* const* d_in, const int* in_sizes, int n_in,
                              void* d_out, int out_size, void* d_ws, size_t ws_size,
                              hipStream_t stream) {
    const int*   erow  = (const int*)  d_in[0];
    const int*   ecol  = (const int*)  d_in[1];
    const float* eval_ = (const float*)d_in[2];
    const float* diag  = (const float*)d_in[3];
    const float* feat  = (const float*)d_in[4];
    const float* W1    = (const float*)d_in[5];
    const float* b1    = (const float*)d_in[6];
    const float* W2    = (const float*)d_in[7];
    const float* b2    = (const float*)d_in[8];
    float* out = (float*)d_out;

    float* Lx    = (float*)d_ws;                       // [N, D] f32
    float* Lxi   = Lx  + (size_t)N_NODES * DFEAT;      // [N, D] f32
    float* inter = Lxi + (size_t)N_NODES * DFEAT;      // [N, D] f32
    _Float16* W1h = (_Float16*)(inter + (size_t)N_NODES * DFEAT); // [D, D] f16
    _Float16* W2h = W1h + DFEAT * DFEAT;                          // [D, D] f16

    // 1) zero the two atomic accumulators (every call: atomics accumulate).
    {
        const int n4 = 2 * N_NODES * DFEAT / 4;
        zero4_kernel<<<(n4 + 255) / 256, 256, 0, stream>>>((float4*)Lx, n4);
    }
    // 2) weights -> f16 (32KB each; stays hot in L2 for the GEMM).
    cvt_w_kernel<<<(DFEAT * DFEAT + 255) / 256, 256, 0, stream>>>(W1, W2, W1h, W2h,
                                                                  DFEAT * DFEAT);
    const int waves     = (N_EDGES + EPW - 1) / EPW;   // 25000
    const int spmm_blks = (waves + 7) / 8;             // 8 waves / block
    // 3) Lx = L @ features
    spmm_kernel<<<spmm_blks, 256, 0, stream>>>(erow, ecol, eval_, feat, Lx);
    // 4) inter = Lx .* features
    {
        const int n4 = N_NODES * DFEAT / 4;
        inter_kernel<<<(n4 + 255) / 256, 256, 0, stream>>>(Lx, feat, inter, n4);
    }
    // 5) Lxi = L @ inter
    spmm_kernel<<<spmm_blks, 256, 0, stream>>>(erow, ecol, eval_, inter, Lxi);
    // 6) out = (Lx + diag.*feat) @ W1^T + b1 + Lxi @ W2^T + b2
    gemm_out_kernel<<<N_NODES / 16, 256, 0, stream>>>(Lx, Lxi, feat, diag,
                                                      W1h, b1, W2h, b2, out);
}
